// projection_net_71554155152058
// MI455X (gfx1250) — matrix operand order repaired and checked by reading, not compiled, not run on hardware
//
#include <hip/hip_runtime.h>

// ---------------------------------------------------------------------------
// CDNA5 (gfx1250) wave32 WMMA bf16 types
// ---------------------------------------------------------------------------
typedef __bf16 bf16_t;
typedef __attribute__((ext_vector_type(16))) __bf16 v16bf;
typedef __attribute__((ext_vector_type(8)))  float  v8f;

// A-matrix (16x32 bf16) per-lane layout: lane half h, element e:
//   K = h*16 + e   -> one contiguous 16-float chunk (caller pre-offsets by half*16)
__device__ __forceinline__ v16bf loadA16(const float* __restrict__ p) {
  v16bf r;
#pragma unroll
  for (int e = 0; e < 8; ++e) r[e] = (bf16_t)p[e];
#pragma unroll
  for (int e = 0; e < 8; ++e) r[e + 8] = (bf16_t)p[e + 8];
  return r;
}

// B-matrix (32x16 bf16) per-lane layout: lane half h, element e: K = h*16 + e
// -> one contiguous 16-float chunk (caller pre-offsets by half*16)
__device__ __forceinline__ v16bf loadB16(const float* __restrict__ p) {
  v16bf r;
#pragma unroll
  for (int e = 0; e < 16; ++e) r[e] = (bf16_t)p[e];
  return r;
}

__device__ __forceinline__ float gelu_exact(float x) {
  return 0.5f * x * (1.0f + erff(x * 0.70710678118654752440f));
}
__device__ __forceinline__ float sigmoidf_(float x) { return 1.0f / (1.0f + expf(-x)); }

// ---------------------------------------------------------------------------
// fp32 -> bf16 conversion (weight pre-pass for the LSTM streams)
// ---------------------------------------------------------------------------
__global__ __launch_bounds__(256) void to_bf16_k(
    const float* __restrict__ in, bf16_t* __restrict__ out, int n)
{
  const int i = blockIdx.x * 256 + threadIdx.x;
  if (i < n) out[i] = (bf16_t)in[i];
}

// ---------------------------------------------------------------------------
// Gated GEMM: out[M,N] = gelu(X @ W1^T + b1) * (X @ W2^T + b2)
// X:[M,K] f32 row-major, W1/W2:[N,K] f32 row-major. K%32==0, M%64==0, N%16==0.
// One wave computes a 64x16 output patch (4 M-tiles), reusing each B-matrix
// load across 4 WMMAs -> 2x less L2 traffic per WMMA vs 1 tile/wave.
// ---------------------------------------------------------------------------
__global__ __launch_bounds__(128) void gated_gemm(
    const float* __restrict__ X,
    const float* __restrict__ W1, const float* __restrict__ b1,
    const float* __restrict__ W2, const float* __restrict__ b2,
    float* __restrict__ out, int M, int K, int N)
{
  const int wave = threadIdx.x >> 5;
  const int lane = threadIdx.x & 31;
  const int half = lane >> 4;
  const int lm   = lane & 15;

  const int ntiles  = N >> 4;
  const int mstrips = M >> 6;                    // 64-row strips
  const int tile = blockIdx.x * 4 + wave;
  if (tile >= mstrips * ntiles) return;          // wave-uniform exit
  const int ts = tile / ntiles;
  const int tn = tile % ntiles;

  const float* xbase = X  + (size_t)(ts * 64 + lm) * K + half * 16;
  const float* w1row = W1 + (size_t)(tn * 16 + lm) * K + half * 16;
  const float* w2row = W2 + (size_t)(tn * 16 + lm) * K + half * 16;

  const v8f vzero = {};
  v8f acc1[4], acc2[4];
#pragma unroll
  for (int mt = 0; mt < 4; ++mt) { acc1[mt] = vzero; acc2[mt] = vzero; }

  for (int k0 = 0; k0 < K; k0 += 32) {
    const v16bf bA = loadB16(w1row + k0);
    const v16bf bB = loadB16(w2row + k0);
#pragma unroll
    for (int mt = 0; mt < 4; ++mt) {
      const v16bf a = loadA16(xbase + (size_t)mt * 16 * K + k0);
      acc1[mt] = __builtin_amdgcn_wmma_f32_16x16x32_bf16(false, a, false, bA, (short)0, acc1[mt], false, false);
      acc2[mt] = __builtin_amdgcn_wmma_f32_16x16x32_bf16(false, a, false, bB, (short)0, acc2[mt], false, false);
    }
  }

  // C/D layout: lane n = lm, VGPR r -> m = r + 8*half
  const float bb1 = b1[tn * 16 + lm];
  const float bb2 = b2[tn * 16 + lm];
#pragma unroll
  for (int mt = 0; mt < 4; ++mt) {
#pragma unroll
    for (int r = 0; r < 8; ++r) {
      const float p1 = acc1[mt][r] + bb1;
      const float p2 = acc2[mt][r] + bb2;
      out[(size_t)(ts * 64 + mt * 16 + r + half * 8) * N + tn * 16 + lm] = gelu_exact(p1) * p2;
    }
  }
}

// ---------------------------------------------------------------------------
// LayerNorm over last dim D (one block per row)
// ---------------------------------------------------------------------------
__global__ __launch_bounds__(256) void layernorm_k(
    const float* __restrict__ x, const float* __restrict__ g,
    const float* __restrict__ bta, float* __restrict__ out, int D)
{
  __shared__ float red[256];
  __shared__ float sMean, sRstd;
  const int row = blockIdx.x, tid = threadIdx.x;
  const float* xr = x + (size_t)row * D;

  float s = 0.0f;
  for (int i = tid; i < D; i += 256) s += xr[i];
  red[tid] = s; __syncthreads();
  for (int off = 128; off > 0; off >>= 1) { if (tid < off) red[tid] += red[tid + off]; __syncthreads(); }
  if (tid == 0) sMean = red[0] / (float)D;
  __syncthreads();
  const float m = sMean;

  float v = 0.0f;
  for (int i = tid; i < D; i += 256) { const float d = xr[i] - m; v += d * d; }
  red[tid] = v; __syncthreads();
  for (int off = 128; off > 0; off >>= 1) { if (tid < off) red[tid] += red[tid + off]; __syncthreads(); }
  if (tid == 0) sRstd = rsqrtf(red[0] / (float)D + 1e-5f);
  __syncthreads();
  const float r = sRstd;

  for (int i = tid; i < D; i += 256)
    out[(size_t)row * D + i] = (xr[i] - m) * r * g[i] + bta[i];
}

// ---------------------------------------------------------------------------
// Vision preprocessing: v0[b,0:1024] = mean(vision[b,1:8,:]); v0[b,1024:2048] = vision[b,0,:]
// ---------------------------------------------------------------------------
__global__ __launch_bounds__(256) void vision_prep(
    const float* __restrict__ vis, float* __restrict__ v0)
{
  const int idx = blockIdx.x * 256 + threadIdx.x;
  if (idx >= 512 * 2048) return;
  const int col = idx & 2047;
  const int b   = idx >> 11;
  const float* vb = vis + (size_t)b * 8 * 1024;
  float r;
  if (col < 1024) {
    float s = 0.0f;
#pragma unroll
    for (int f = 1; f < 8; ++f) s += vb[f * 1024 + col];
    r = s * (1.0f / 7.0f);
  } else {
    r = vb[col - 1024];
  }
  v0[idx] = r;
}

// ---------------------------------------------------------------------------
// Fused LSTM (input projection + recurrence + gates), torch gate order i,f,g,o.
// Grid: 32 blocks x 128 threads. Block owns 16 batch rows, wave owns 4 h-tile
// groups (ht = wave + 4j) x 4 gates = 16 WMMA f32 accumulators. h carried via
// padded bf16 LDS (read back as WMMA A-matrices); c stays in VGPRs. Weights
// come pre-converted to bf16 (direct 32B-aligned v16bf B-matrix loads).
// hs out: [512, 64*256] with h fastest (t*256 + h).
// ---------------------------------------------------------------------------
__global__ __launch_bounds__(128) void lstm_kernel(
    const float*  __restrict__ x,     // [512,64,768] f32
    const bf16_t* __restrict__ Wih,   // [1024,768] bf16
    const bf16_t* __restrict__ Whh,   // [1024,256] bf16
    const float* __restrict__ bih, const float* __restrict__ bhh,
    float* __restrict__ hs)
{
  constexpr int HPAD = 264;                 // bf16 row stride (padding vs LDS banks)
  __shared__ bf16_t hlds[16 * HPAD];
  const int tid  = threadIdx.x;
  const int wave = tid >> 5;
  const int lane = tid & 31;
  const int half = lane >> 4;
  const int lm   = lane & 15;
  const int b0   = blockIdx.x * 16;

  for (int i = tid; i < 16 * HPAD; i += 128) hlds[i] = (bf16_t)0.0f;

  const v8f vzero = {};
  v8f cst[4];
  float bias[4][4];
#pragma unroll
  for (int j = 0; j < 4; ++j) {
    cst[j] = vzero;
    const int ht = wave + 4 * j;
#pragma unroll
    for (int g = 0; g < 4; ++g) {
      const int n = g * 256 + ht * 16 + lm;
      bias[j][g] = bih[n] + bhh[n];
    }
  }
  __syncthreads();

  for (int t = 0; t < 64; ++t) {
    v8f acc[4][4];
#pragma unroll
    for (int j = 0; j < 4; ++j)
#pragma unroll
      for (int g = 0; g < 4; ++g) acc[j][g] = vzero;

    // ---- input projection: x_t @ Wih^T (K = 768) ----
    const float* xt = x + ((size_t)(b0 + lm) * 64 + t) * 768 + half * 16;
    for (int kc = 0; kc < 24; ++kc) {
      const v16bf a = loadA16(xt + kc * 32);
#pragma unroll
      for (int j = 0; j < 4; ++j) {
        const int ht = wave + 4 * j;
#pragma unroll
        for (int g = 0; g < 4; ++g) {
          const int n = g * 256 + ht * 16 + lm;
          const v16bf bm = *reinterpret_cast<const v16bf*>(Wih + (size_t)n * 768 + kc * 32 + half * 16);
          acc[j][g] = __builtin_amdgcn_wmma_f32_16x16x32_bf16(false, a, false, bm, (short)0, acc[j][g], false, false);
        }
      }
    }
    // ---- recurrence: h_{t-1} @ Whh^T (K = 256), A from LDS ----
    for (int kc = 0; kc < 8; ++kc) {
      const bf16_t* hp = hlds + lm * HPAD + kc * 32 + half * 16;
      v16bf a;
#pragma unroll
      for (int e = 0; e < 8; ++e) a[e] = hp[e];
#pragma unroll
      for (int e = 0; e < 8; ++e) a[e + 8] = hp[e + 8];
#pragma unroll
      for (int j = 0; j < 4; ++j) {
        const int ht = wave + 4 * j;
#pragma unroll
        for (int g = 0; g < 4; ++g) {
          const int n = g * 256 + ht * 16 + lm;
          const v16bf bm = *reinterpret_cast<const v16bf*>(Whh + (size_t)n * 256 + kc * 32 + half * 16);
          acc[j][g] = __builtin_amdgcn_wmma_f32_16x16x32_bf16(false, a, false, bm, (short)0, acc[j][g], false, false);
        }
      }
    }
    __syncthreads();   // all reads of h_{t-1} complete before overwriting

    // ---- gates + state update + write h ----
#pragma unroll
    for (int j = 0; j < 4; ++j) {
      const int ht = wave + 4 * j;
#pragma unroll
      for (int r = 0; r < 8; ++r) {
        const float iv = sigmoidf_(acc[j][0][r] + bias[j][0]);
        const float fv = sigmoidf_(acc[j][1][r] + bias[j][1]);
        const float gv = tanhf(acc[j][2][r] + bias[j][2]);
        const float ov = sigmoidf_(acc[j][3][r] + bias[j][3]);
        const float c  = fv * cst[j][r] + iv * gv;
        const float h  = ov * tanhf(c);
        cst[j][r] = c;
        const int m    = r + half * 8;
        const int hcol = ht * 16 + lm;
        hs[(size_t)(b0 + m) * 16384 + t * 256 + hcol] = h;
        hlds[m * HPAD + hcol] = (bf16_t)h;
      }
    }
    __syncthreads();   // h_t visible before next step reads
  }
}

// ---------------------------------------------------------------------------
// Fusion: l2norm + 3 cross-CASA + self-CASA(+residual) + fold + gated head.
// One block per batch element (tiny FLOPs -> plain fp32 VALU).
// ---------------------------------------------------------------------------
struct CasaP { const float *Wk, *bk, *Wv, *bv, *Wq, *bq; };

__global__ __launch_bounds__(256) void fusion_kernel(
    const float* __restrict__ afeat, const float* __restrict__ vfeat,
    const float* __restrict__ sfeat,
    CasaP pZA, CasaP pZV, CasaP pZT, CasaP pSA,
    const float* __restrict__ pW1, const float* __restrict__ pb1,
    const float* __restrict__ pW2, const float* __restrict__ pb2,
    float* __restrict__ prob)
{
  const int b = blockIdx.x, tid = threadIdx.x;
  __shared__ float raw[3][512];
  __shared__ float z[1536];          // normalized concat, [48][32]
  __shared__ float K2[48 * 32];
  __shared__ float V2[48 * 32];
  __shared__ float feat[48 * 32];
  __shared__ float feat2[48 * 32];
  __shared__ float red[256];
  __shared__ float red4[4][256];
  __shared__ float rnorm[3];

  const float* srcs[3] = { afeat + (size_t)b * 512, vfeat + (size_t)b * 512, sfeat + (size_t)b * 512 };
  for (int s = 0; s < 3; ++s)
    for (int i = tid; i < 512; i += 256) raw[s][i] = srcs[s][i];
  __syncthreads();

  for (int s = 0; s < 3; ++s) {
    float p = 0.0f;
    for (int i = tid; i < 512; i += 256) { const float xx = raw[s][i]; p += xx * xx; }
    red[tid] = p; __syncthreads();
    for (int off = 128; off > 0; off >>= 1) { if (tid < off) red[tid] += red[tid + off]; __syncthreads(); }
    if (tid == 0) rnorm[s] = 1.0f / fmaxf(sqrtf(red[0]), 1e-12f);
    __syncthreads();
  }
  for (int s = 0; s < 3; ++s)
    for (int i = tid; i < 512; i += 256) z[s * 512 + i] = raw[s][i] * rnorm[s];
  for (int i = tid; i < 1536; i += 256) feat[i] = 0.0f;
  __syncthreads();

  // three cross attentions: f1 = z [48,32], f2 = raw feat s [16,32]
  for (int s = 0; s < 3; ++s) {
    const CasaP P = (s == 0) ? pZA : ((s == 1) ? pZV : pZT);
    for (int idx = tid; idx < 512; idx += 256) {
      const int i = idx >> 5, d = idx & 31;
      const float* f2r = raw[s] + i * 32;
      const float* wk = P.Wk + d * 32;
      const float* wv = P.Wv + d * 32;
      float sk = P.bk[d], sv = P.bv[d];
      for (int k = 0; k < 32; ++k) { sk += f2r[k] * wk[k]; sv += f2r[k] * wv[k]; }
      K2[idx] = sk; V2[idx] = sv;
    }
    __syncthreads();
    if (tid < 48) {
      const int i = tid;
      const float* f1r = z + i * 32;
      float q[32];
      for (int d = 0; d < 32; ++d) {
        const float* wq = P.Wq + d * 32;
        float s2 = P.bq[d];
        for (int k = 0; k < 32; ++k) s2 += f1r[k] * wq[k];
        q[d] = s2;
      }
      float lg[16]; float mx = -1e30f;
      for (int j = 0; j < 16; ++j) {
        const float* kr = K2 + j * 32;
        float s2 = 0.0f;
        for (int k = 0; k < 32; ++k) s2 += q[k] * kr[k];
        lg[j] = s2; mx = fmaxf(mx, s2);
      }
      float den = 0.0f;
      for (int j = 0; j < 16; ++j) { lg[j] = expf(lg[j] - mx); den += lg[j]; }
      const float inv = 1.0f / den;
      for (int d = 0; d < 32; ++d) {
        float o = 0.0f;
        for (int j = 0; j < 16; ++j) o += lg[j] * V2[j * 32 + d];
        feat[i * 32 + d] += o * inv;
      }
    }
    __syncthreads();
  }

  // self attention on feat [48,32] with residual
  for (int idx = tid; idx < 1536; idx += 256) {
    const int i = idx >> 5, d = idx & 31;
    const float* fr = feat + i * 32;
    const float* wk = pSA.Wk + d * 32;
    const float* wv = pSA.Wv + d * 32;
    float sk = pSA.bk[d], sv = pSA.bv[d];
    for (int k = 0; k < 32; ++k) { sk += fr[k] * wk[k]; sv += fr[k] * wv[k]; }
    K2[idx] = sk; V2[idx] = sv;
  }
  __syncthreads();
  if (tid < 48) {
    const int i = tid;
    const float* f1r = feat + i * 32;
    float q[32];
    for (int d = 0; d < 32; ++d) {
      const float* wq = pSA.Wq + d * 32;
      float s2 = pSA.bq[d];
      for (int k = 0; k < 32; ++k) s2 += f1r[k] * wq[k];
      q[d] = s2;
    }
    float lg[48]; float mx = -1e30f;
    for (int j = 0; j < 48; ++j) {
      const float* kr = K2 + j * 32;
      float s2 = 0.0f;
      for (int k = 0; k < 32; ++k) s2 += q[k] * kr[k];
      lg[j] = s2; mx = fmaxf(mx, s2);
    }
    float den = 0.0f;
    for (int j = 0; j < 48; ++j) { lg[j] = expf(lg[j] - mx); den += lg[j]; }
    const float inv = 1.0f / den;
    for (int d = 0; d < 32; ++d) {
      float o = 0.0f;
      for (int j = 0; j < 48; ++j) o += lg[j] * V2[j * 32 + d];
      feat2[i * 32 + d] = o * inv + f1r[d];
    }
  }
  __syncthreads();

  // fold (f1+f2+f3) -> [512], then gated head to 2 outputs
  float p10 = 0.0f, p11 = 0.0f, p20 = 0.0f, p21 = 0.0f;
  for (int idx = tid; idx < 512; idx += 256) {
    const float fo = feat2[idx] + feat2[512 + idx] + feat2[1024 + idx];
    p10 += fo * pW1[idx];
    p11 += fo * pW1[512 + idx];
    p20 += fo * pW2[idx];
    p21 += fo * pW2[512 + idx];
  }
  red4[0][tid] = p10; red4[1][tid] = p11; red4[2][tid] = p20; red4[3][tid] = p21;
  __syncthreads();
  for (int off = 128; off > 0; off >>= 1) {
    if (tid < off) {
#pragma unroll
      for (int q2 = 0; q2 < 4; ++q2) red4[q2][tid] += red4[q2][tid + off];
    }
    __syncthreads();
  }
  if (tid == 0) {
    const float h10 = red4[0][0] + pb1[0];
    const float h11 = red4[1][0] + pb1[1];
    const float h20 = red4[2][0] + pb2[0];
    const float h21 = red4[3][0] + pb2[1];
    prob[(size_t)b * 2 + 0] = gelu_exact(h10) * h20;
    prob[(size_t)b * 2 + 1] = gelu_exact(h11) * h21;
  }
}

// ---------------------------------------------------------------------------
// Host launch
// ---------------------------------------------------------------------------
extern "C" void kernel_launch(void* const* d_in, const int* in_sizes, int n_in,
                              void* d_out, int out_size, void* d_ws, size_t ws_size,
                              hipStream_t stream) {
  (void)in_sizes; (void)n_in; (void)out_size; (void)ws_size;

  const float* audio  = (const float*)d_in[0];   // [512,2048]
  const float* vision = (const float*)d_in[1];   // [512,8,1024]
  const float* sub    = (const float*)d_in[2];   // [512,64,768]
  // audio branch params
  const float* aW1 = (const float*)d_in[3];  const float* ab1 = (const float*)d_in[4];
  const float* aW2 = (const float*)d_in[5];  const float* ab2 = (const float*)d_in[6];
  const float* aW3 = (const float*)d_in[7];  const float* ab3 = (const float*)d_in[8];
  const float* aW4 = (const float*)d_in[9];  const float* ab4 = (const float*)d_in[10];
  const float* aLNg = (const float*)d_in[11]; const float* aLNb = (const float*)d_in[12];
  // vision branch params
  const float* vW1 = (const float*)d_in[13]; const float* vb1 = (const float*)d_in[14];
  const float* vW2 = (const float*)d_in[15]; const float* vb2 = (const float*)d_in[16];
  const float* vW3 = (const float*)d_in[17]; const float* vb3 = (const float*)d_in[18];
  const float* vW4 = (const float*)d_in[19]; const float* vb4 = (const float*)d_in[20];
  const float* vLNg = (const float*)d_in[21]; const float* vLNb = (const float*)d_in[22];
  // lstm
  const float* Wih = (const float*)d_in[23]; const float* Whh = (const float*)d_in[24];
  const float* bih = (const float*)d_in[25]; const float* bhh = (const float*)d_in[26];
  // subtitle branch params
  const float* sW1 = (const float*)d_in[27]; const float* sb1 = (const float*)d_in[28];
  const float* sW2 = (const float*)d_in[29]; const float* sb2 = (const float*)d_in[30];
  const float* sW3 = (const float*)d_in[31]; const float* sb3 = (const float*)d_in[32];
  const float* sW4 = (const float*)d_in[33]; const float* sb4 = (const float*)d_in[34];
  const float* sLNg = (const float*)d_in[35]; const float* sLNb = (const float*)d_in[36];
  // CASA params
  CasaP ZA = { (const float*)d_in[37], (const float*)d_in[38], (const float*)d_in[39],
               (const float*)d_in[40], (const float*)d_in[41], (const float*)d_in[42] };
  CasaP ZV = { (const float*)d_in[43], (const float*)d_in[44], (const float*)d_in[45],
               (const float*)d_in[46], (const float*)d_in[47], (const float*)d_in[48] };
  CasaP ZT = { (const float*)d_in[49], (const float*)d_in[50], (const float*)d_in[51],
               (const float*)d_in[52], (const float*)d_in[53], (const float*)d_in[54] };
  CasaP SA = { (const float*)d_in[55], (const float*)d_in[56], (const float*)d_in[57],
               (const float*)d_in[58], (const float*)d_in[59], (const float*)d_in[60] };
  // head
  const float* pW1 = (const float*)d_in[61]; const float* pb1 = (const float*)d_in[62];
  const float* pW2 = (const float*)d_in[63]; const float* pb2 = (const float*)d_in[64];

  // workspace layout
  float* ws   = (float*)d_ws;
  float* a1   = ws;                       // 512*1024
  float* buf0 = a1 + 512 * 1024;          // 512*512
  float* buf1 = buf0 + 512 * 512;         // 512*512
  float* v0   = buf1 + 512 * 512;         // 512*2048
  float* hsb  = v0 + 512 * 2048;          // 512*16384
  bf16_t* Wihb = (bf16_t*)(hsb + 512 * 16384);   // 1024*768 bf16
  bf16_t* Whhb = Wihb + 1024 * 768;              // 1024*256 bf16

  float* out        = (float*)d_out;
  float* audio_feat = out;                       // [512,512]
  float* vis_feat   = out + 512 * 512;           // [512,512]
  float* sub_feat   = out + 2 * 512 * 512;       // [512,512]
  float* prob       = out + 3 * 512 * 512;       // [512,2]

  // LSTM weight bf16 pre-pass (overlaps audio/vision branches on the stream)
  to_bf16_k<<<(1024 * 768) / 256, 256, 0, stream>>>(Wih, Wihb, 1024 * 768);
  to_bf16_k<<<(1024 * 256) / 256, 256, 0, stream>>>(Whh, Whhb, 1024 * 256);

  // audio branch  (wave patch = 64x16 -> waves = (M/64)*(N/16), 4 waves/block)
  gated_gemm<<<(8 * 64) / 4, 128, 0, stream>>>(audio, aW1, ab1, aW2, ab2, a1, 512, 2048, 1024);
  gated_gemm<<<(8 * 32) / 4, 128, 0, stream>>>(a1, aW3, ab3, aW4, ab4, buf0, 512, 1024, 512);
  layernorm_k<<<512, 256, 0, stream>>>(buf0, aLNg, aLNb, audio_feat, 512);

  // vision branch
  vision_prep<<<(512 * 2048) / 256, 256, 0, stream>>>(vision, v0);
  gated_gemm<<<(8 * 32) / 4, 128, 0, stream>>>(v0, vW1, vb1, vW2, vb2, buf0, 512, 2048, 512);
  gated_gemm<<<(8 * 32) / 4, 128, 0, stream>>>(buf0, vW3, vb3, vW4, vb4, buf1, 512, 512, 512);
  layernorm_k<<<512, 256, 0, stream>>>(buf1, vLNg, vLNb, vis_feat, 512);

  // subtitle branch
  lstm_kernel<<<32, 128, 0, stream>>>(sub, Wihb, Whhb, bih, bhh, hsb);
  gated_gemm<<<(8 * 32) / 4, 128, 0, stream>>>(hsb, sW1, sb1, sW2, sb2, buf0, 512, 16384, 512);
  gated_gemm<<<(8 * 32) / 4, 128, 0, stream>>>(buf0, sW3, sb3, sW4, sb4, buf1, 512, 512, 512);
  layernorm_k<<<512, 256, 0, stream>>>(buf1, sLNg, sLNb, sub_feat, 512);

  // fusion + head
  fusion_kernel<<<512, 256, 0, stream>>>(audio_feat, vis_feat, sub_feat,
                                         ZA, ZV, ZT, SA, pW1, pb1, pW2, pb2, prob);
}
